// CustomMambaBlock_10196252360760
// MI455X (gfx1250) — matrix operand order repaired and checked by reading, not compiled, not run on hardware
//
#include <hip/hip_runtime.h>
#include <stdint.h>

// Problem constants (match reference)
static constexpr int B_ = 2, L_ = 1024, H_ = 768, E_ = 1536, N_ = 16, R_ = 48, K_ = 3;
static constexpr int LP_ = L_ + 4;   // padded sequence length (2 guard rows each side)

typedef __attribute__((ext_vector_type(16))) __bf16    v16bf;
typedef __attribute__((ext_vector_type(8)))  float     v8f;
typedef __attribute__((ext_vector_type(8)))  uint32_t  v8u;
typedef __attribute__((ext_vector_type(4)))  uint32_t  v4u;

__device__ __forceinline__ unsigned short f2bf(float f) {
  uint32_t u = __builtin_bit_cast(uint32_t, f);
  u += 0x7fffu + ((u >> 16) & 1u);          // round-to-nearest-even
  return (unsigned short)(u >> 16);
}

// ---- WMMA fragment loaders (bf16 16x16x32), branch-free --------------------
// A fragment (16M x 32K): lane&15 = row M, half = lane>>4.
// Lane data = rp[kBase+half*8 .. +7] ++ rp[kBase+16+half*8 .. +7]  (2x b128)
__device__ __forceinline__ v16bf load_a(const unsigned short* rp, int kBase, int half) {
  v4u lo = *(const v4u*)(rp + kBase + half * 8);
  v4u hi = *(const v4u*)(rp + kBase + 16 + half * 8);
  v8u u;
  u[0] = lo[0]; u[1] = lo[1]; u[2] = lo[2]; u[3] = lo[3];
  u[4] = hi[0]; u[5] = hi[1]; u[6] = hi[2]; u[7] = hi[3];
  return __builtin_bit_cast(v16bf, u);
}
// B fragment (32K x 16N) from row-major W[N][K]: lane&15 = col N.
// Lane data = rp[kBase+half*8 .. +7] ++ rp[kBase+16+half*8 .. +7]  (2x b128)
__device__ __forceinline__ v16bf load_b(const unsigned short* rp, int kBase, int half) {
  const unsigned short* q = rp + kBase + half * 8;
  v4u lo = *(const v4u*)(q);
  v4u hi = *(const v4u*)(q + 16);
  v8u u;
  u[0] = lo[0]; u[1] = lo[1]; u[2] = lo[2]; u[3] = lo[3];
  u[4] = hi[0]; u[5] = hi[1]; u[6] = hi[2]; u[7] = hi[3];
  return __builtin_bit_cast(v16bf, u);
}

// ---- Generic WMMA GEMM: out = epilogue(A[M,K] @ W[N,K]^T + bias) -----------
// MODE 0: outBF(padded xy layout) = bf16(v), outF = silu(v)   (in-projection)
// MODE 2: outF = v                                             (dbc)
// MODE 3: outF = softplus(v)                                   (delta)
// MODE 4: outF = resid + v                                     (out-projection)
template <int NT, int MODE>
__global__ void k_wmma_gemm(const unsigned short* __restrict__ A, int lda,
                            const unsigned short* __restrict__ W, int ldb,
                            int M, int N, int K, const float* __restrict__ bias,
                            float* __restrict__ outF,
                            unsigned short* __restrict__ outBF,
                            const float* __restrict__ resid) {
  const int lane = threadIdx.x & 31;
  const int wave = threadIdx.x >> 5;
  const int groupsN = (N >> 4) / NT;
  const int total = (M >> 4) * groupsN;
  const int tile = blockIdx.x * (blockDim.x >> 5) + wave;
  if (tile >= total) return;                 // wave-uniform: EXEC stays all-ones
  const int tm = tile / groupsN, tg = tile - tm * groupsN;
  const int half = lane >> 4, l16 = lane & 15;
  const unsigned short* arow = A + (size_t)(tm * 16 + l16) * lda;
  const unsigned short* brow[NT];
  v8f c[NT] = {};
#pragma unroll
  for (int t = 0; t < NT; ++t)
    brow[t] = W + (size_t)((tg * NT + t) * 16 + l16) * ldb;

  for (int kk = 0; kk < K; kk += 32) {
    __builtin_prefetch(arow + kk + 256, 0, 3);
    v16bf a = load_a(arow, kk, half);
#pragma unroll
    for (int t = 0; t < NT; ++t) {
      v16bf b = load_b(brow[t], kk, half);
      c[t] = __builtin_amdgcn_wmma_f32_16x16x32_bf16(false, a, false, b, (short)0,
                                                     c[t], false, false);
    }
  }
#pragma unroll
  for (int t = 0; t < NT; ++t) {
    const int n = (tg * NT + t) * 16 + l16;
    const float bv = bias ? bias[n] : 0.f;
#pragma unroll
    for (int r = 0; r < 8; ++r) {
      const int m = tm * 16 + half * 8 + r;  // C layout: VGPR r -> M = half*8 + r
      float v = c[t][r] + bv;
      const size_t o = (size_t)m * N + n;
      if (MODE == 0) {
        const int b = m / L_, l = m - b * L_;
        outBF[(size_t)(b * LP_ + 2 + l) * N + n] = f2bf(v);
        outF[o] = v / (1.f + __expf(-v));    // z = silu(xy)
      } else if (MODE == 2) {
        outF[o] = v;
      } else if (MODE == 3) {
        outF[o] = (v > 20.f) ? v : __logf(1.f + __expf(v));   // softplus
      } else {
        outF[o] = resid[o] + v;
      }
    }
  }
}

// ---- Causal conv: 3 shift-accumulated WMMA GEMMs over padded xy ------------
// dir=0 (fwd): y[l] = sum_s xy[l+s-2] @ Ws^T ; dir=1 (bwd): xy[l+2-s]
__global__ void k_wmma_conv(const unsigned short* __restrict__ xyp, // [B][LP][E]
                            const unsigned short* __restrict__ cwT, // [k][o][i]
                            const float* __restrict__ cb, int dir,
                            unsigned short* __restrict__ uBF,
                            float* __restrict__ uF) {
  const int K = E_;
  const int lane = threadIdx.x & 31;
  const int wave = threadIdx.x >> 5;
  const int groupsN = (E_ >> 4) / 2;         // NT = 2
  const int total = ((B_ * L_) >> 4) * groupsN;
  const int tile = blockIdx.x * (blockDim.x >> 5) + wave;
  if (tile >= total) return;
  const int tm = tile / groupsN, tg = tile - tm * groupsN;
  const int half = lane >> 4, l16 = lane & 15;
  const int gm = tm * 16 + l16;
  const int b = gm / L_, l = gm - b * L_;
  const unsigned short* w0 = cwT + (size_t)(tg * 32 + l16) * K;
  const unsigned short* w1 = w0 + (size_t)16 * K;
  v8f c0 = {}, c1 = {};
#pragma unroll
  for (int s = 0; s < 3; ++s) {
    const int sl = dir ? (l + 2 - s) : (l + s - 2);       // in [-2, L+1]
    const unsigned short* arow = xyp + (size_t)(b * LP_ + 2 + sl) * E_;
    const unsigned short* b0 = w0 + (size_t)s * E_ * E_;
    const unsigned short* b1 = w1 + (size_t)s * E_ * E_;
    for (int kk = 0; kk < K; kk += 32) {
      v16bf a  = load_a(arow, kk, half);
      v16bf f0 = load_b(b0, kk, half);
      v16bf f1 = load_b(b1, kk, half);
      c0 = __builtin_amdgcn_wmma_f32_16x16x32_bf16(false, a, false, f0, (short)0,
                                                   c0, false, false);
      c1 = __builtin_amdgcn_wmma_f32_16x16x32_bf16(false, a, false, f1, (short)0,
                                                   c1, false, false);
    }
  }
  const float* cbp = cb + tg * 32;
#pragma unroll
  for (int t = 0; t < 2; ++t) {
    const int n = tg * 32 + t * 16 + l16;
    const float bv = cbp[t * 16 + l16];
    const v8f& c = t ? c1 : c0;
#pragma unroll
    for (int r = 0; r < 8; ++r) {
      const int m = tm * 16 + half * 8 + r;
      float v = c[r] + bv;
      v = v / (1.f + __expf(-v));            // silu
      const size_t o = (size_t)m * E_ + n;
      uBF[o] = f2bf(v);
      uF[o]  = v;
    }
  }
}

// ---- SSM selective scan: 1 thread per (b, e); h[N] in registers ------------
__global__ void k_ssm(const float* __restrict__ delta, const float* __restrict__ dbc,
                      const float* __restrict__ u, const float* __restrict__ A_log,
                      const float* __restrict__ Dp, float* __restrict__ y, int add) {
  const int t = blockIdx.x * blockDim.x + threadIdx.x;
  if (t >= B_ * E_) return;
  const int b = t / E_, e = t - b * E_;
  float Aa[N_];
#pragma unroll
  for (int n = 0; n < N_; ++n) Aa[n] = -__expf(A_log[e * N_ + n]);
  const float D = Dp[e];
  float h[N_];
#pragma unroll
  for (int n = 0; n < N_; ++n) h[n] = 0.f;
  const float* dB   = delta + (size_t)b * L_ * E_ + e;
  const float* uB   = u     + (size_t)b * L_ * E_ + e;
  const float* dbcB = dbc   + (size_t)b * L_ * (R_ + 2 * N_);
  float* yB         = y     + (size_t)b * L_ * E_ + e;
  for (int l = 0; l < L_; ++l) {
    const float d  = dB[(size_t)l * E_];
    const float uu = uB[(size_t)l * E_];
    const float du = d * uu;
    const float* row = dbcB + (size_t)l * (R_ + 2 * N_);
    float acc = 0.f;
#pragma unroll
    for (int n = 0; n < N_; ++n) {
      const float dA = __expf(d * Aa[n]);
      h[n] = dA * h[n] + du * row[R_ + n];   // Bm
      acc += h[n] * row[R_ + N_ + n];        // Cm
    }
    const float val = acc + D * uu;
    if (add) yB[(size_t)l * E_] += val;
    else     yB[(size_t)l * E_] = val;
  }
}

// ---- LayerNorm over whole (L,H) per batch: deterministic two-pass ----------
__global__ void k_ln_part(const float* __restrict__ x, float* __restrict__ part) {
  __shared__ float ss[256], sq[256];
  const int b = blockIdx.y;
  const size_t LH = (size_t)L_ * H_;
  const float* xb = x + (size_t)b * LH;
  float s = 0.f, q = 0.f;
  for (size_t i = (size_t)blockIdx.x * 256 + threadIdx.x; i < LH;
       i += (size_t)gridDim.x * 256) {
    const float v = xb[i]; s += v; q += v * v;
  }
  ss[threadIdx.x] = s; sq[threadIdx.x] = q;
  __syncthreads();
  for (int st = 128; st > 0; st >>= 1) {
    if ((int)threadIdx.x < st) {
      ss[threadIdx.x] += ss[threadIdx.x + st];
      sq[threadIdx.x] += sq[threadIdx.x + st];
    }
    __syncthreads();
  }
  if (threadIdx.x == 0) {
    part[(b * gridDim.x + blockIdx.x) * 2 + 0] = ss[0];
    part[(b * gridDim.x + blockIdx.x) * 2 + 1] = sq[0];
  }
}
__global__ void k_ln_final(const float* __restrict__ part, float* __restrict__ stats,
                           int PB) {
  const int b = threadIdx.x;
  if (b < B_) {
    float s = 0.f, q = 0.f;
    for (int i = 0; i < PB; ++i) {
      s += part[(b * PB + i) * 2];
      q += part[(b * PB + i) * 2 + 1];
    }
    const float inv = 1.f / ((float)L_ * (float)H_);
    const float mu = s * inv;
    const float var = q * inv - mu * mu;
    stats[2 * b]     = mu;
    stats[2 * b + 1] = rsqrtf(var + 1e-5f);
  }
}
__global__ void k_ln_apply(const float* __restrict__ x, const float* __restrict__ lnw,
                           const float* __restrict__ lnb, const float* __restrict__ stats,
                           unsigned short* __restrict__ x1) {
  const size_t LH = (size_t)L_ * H_;
  const size_t i = (size_t)blockIdx.x * blockDim.x + threadIdx.x;
  if (i >= (size_t)B_ * LH) return;
  const int b = (int)(i / LH);
  const size_t j = i - (size_t)b * LH;
  const float v = (x[i] - stats[2 * b]) * stats[2 * b + 1] * lnw[j] + lnb[j];
  x1[i] = f2bf(v);
}

// ---- Elementwise converters / padding --------------------------------------
__global__ void k_cvt(const float* __restrict__ src, unsigned short* __restrict__ dst,
                      size_t n) {
  const size_t i = (size_t)blockIdx.x * blockDim.x + threadIdx.x;
  if (i < n) dst[i] = f2bf(src[i]);
}
// conv weights (E_out, E_in, K) -> [k][o][i] contiguous bf16
__global__ void k_cvt_convw(const float* __restrict__ src,
                            unsigned short* __restrict__ dst) {
  const size_t n = (size_t)E_ * E_ * K_;
  const size_t i = (size_t)blockIdx.x * blockDim.x + threadIdx.x;
  if (i >= n) return;
  const int k = (int)(i % K_);
  const size_t rem = i / K_;                 // = o*E + in
  dst[(size_t)k * E_ * E_ + rem] = f2bf(src[i]);
}
// first R cols of dbc (B*L, R+2N) -> bf16 (B*L, 64), zero-padded K 48->64
__global__ void k_cvt_dbc48(const float* __restrict__ dbc,
                            unsigned short* __restrict__ dst) {
  const size_t n = (size_t)B_ * L_ * 64;
  const size_t i = (size_t)blockIdx.x * blockDim.x + threadIdx.x;
  if (i >= n) return;
  const size_t m = i >> 6;
  const int k = (int)(i & 63);
  dst[i] = (k < R_) ? f2bf(dbc[m * (R_ + 2 * N_) + k]) : (unsigned short)0;
}
// dt_w (E, R) -> bf16 (E, 64), zero-padded K 48->64
__global__ void k_cvt_dtw(const float* __restrict__ w,
                          unsigned short* __restrict__ dst) {
  const size_t n = (size_t)E_ * 64;
  const size_t i = (size_t)blockIdx.x * blockDim.x + threadIdx.x;
  if (i >= n) return;
  const size_t o = i >> 6;
  const int k = (int)(i & 63);
  dst[i] = (k < R_) ? f2bf(w[o * R_ + k]) : (unsigned short)0;
}
// zero the 2+2 guard rows of padded xy per batch
__global__ void k_zero_guard(unsigned short* __restrict__ xyp) {
  const size_t n = (size_t)B_ * 4 * E_;
  const size_t i = (size_t)blockIdx.x * blockDim.x + threadIdx.x;
  if (i >= n) return;
  const int e = (int)(i % E_);
  const int r = (int)(i / E_);
  const int b = r >> 2, rr = r & 3;
  const int row = b * LP_ + (rr < 2 ? rr : L_ + rr);   // rows 0,1,L+2,L+3
  xyp[(size_t)row * E_ + e] = 0;
}
__global__ void k_combine(const float* __restrict__ y, const float* __restrict__ z,
                          unsigned short* __restrict__ s, size_t n) {
  const size_t i = (size_t)blockIdx.x * blockDim.x + threadIdx.x;
  if (i < n) s[i] = f2bf(y[i] * z[i]);
}

// ---------------------------------------------------------------------------
extern "C" void kernel_launch(void* const* d_in, const int* in_sizes, int n_in,
                              void* d_out, int out_size, void* d_ws, size_t ws_size,
                              hipStream_t stream) {
  const float* x      = (const float*)d_in[0];
  const float* ln_w   = (const float*)d_in[1];
  const float* ln_b   = (const float*)d_in[2];
  const float* in_w   = (const float*)d_in[3];
  const float* in_b   = (const float*)d_in[4];
  const float* fwd_cw = (const float*)d_in[5];
  const float* fwd_cb = (const float*)d_in[6];
  const float* bwd_cw = (const float*)d_in[7];
  const float* bwd_cb = (const float*)d_in[8];
  const float* dbc_w  = (const float*)d_in[9];
  const float* dt_w   = (const float*)d_in[10];
  const float* dt_b   = (const float*)d_in[11];
  const float* A_log  = (const float*)d_in[12];
  const float* Dp     = (const float*)d_in[13];
  const float* out_w  = (const float*)d_in[14];
  const float* out_b  = (const float*)d_in[15];
  float* out = (float*)d_out;

  const size_t M = (size_t)B_ * L_;          // 2048 token rows
  char* base = (char*)d_ws;
  size_t off = 0;
  auto alloc = [&](size_t bytes) -> void* {
    void* p = base + off;
    off = (off + bytes + 255) & ~(size_t)255;
    return p;
  };
  float*          ln_part  = (float*)alloc(2ull * 64 * 2 * 4);
  float*          ln_stats = (float*)alloc(4 * 4);
  unsigned short* x1_bf    = (unsigned short*)alloc(M * H_ * 2);
  unsigned short* inw_bf   = (unsigned short*)alloc((size_t)E_ * H_ * 2);
  unsigned short* xyp_bf   = (unsigned short*)alloc((size_t)B_ * LP_ * E_ * 2);
  float*          z        = (float*)alloc(M * E_ * 4);
  unsigned short* fcwT     = (unsigned short*)alloc((size_t)K_ * E_ * E_ * 2);
  unsigned short* bcwT     = (unsigned short*)alloc((size_t)K_ * E_ * E_ * 2);
  unsigned short* dbcw_bf  = (unsigned short*)alloc((size_t)(R_ + 2 * N_) * E_ * 2);
  unsigned short* dtw_bf   = (unsigned short*)alloc((size_t)E_ * 64 * 2);
  unsigned short* outw_bf  = (unsigned short*)alloc((size_t)H_ * E_ * 2);
  unsigned short* u_bf     = (unsigned short*)alloc(M * E_ * 2);
  float*          u_f      = (float*)alloc(M * E_ * 4);
  float*          dbc      = (float*)alloc(M * (R_ + 2 * N_) * 4);
  unsigned short* dbc48_bf = (unsigned short*)alloc(M * 64 * 2);
  float*          delta    = (float*)alloc(M * E_ * 4);
  float*          ysum     = (float*)alloc(M * E_ * 4);
  unsigned short* s_bf     = (unsigned short*)alloc(M * E_ * 2);
  (void)ws_size; (void)n_in; (void)in_sizes; (void)out_size;

  auto eb = [](size_t n) { return (unsigned)((n + 255) / 256); };

  // 1) LayerNorm (per-batch, deterministic)
  k_ln_part<<<dim3(64, B_), 256, 0, stream>>>(x, ln_part);
  k_ln_final<<<1, 32, 0, stream>>>(ln_part, ln_stats, 64);
  k_ln_apply<<<eb(M * H_), 256, 0, stream>>>(x, ln_w, ln_b, ln_stats, x1_bf);

  // 2) weights -> bf16 ; zero conv guard rows
  k_cvt<<<eb((size_t)E_ * H_), 256, 0, stream>>>(in_w, inw_bf, (size_t)E_ * H_);
  k_cvt_convw<<<eb((size_t)E_ * E_ * K_), 256, 0, stream>>>(fwd_cw, fcwT);
  k_cvt_convw<<<eb((size_t)E_ * E_ * K_), 256, 0, stream>>>(bwd_cw, bcwT);
  k_cvt<<<eb((size_t)(R_ + 2 * N_) * E_), 256, 0, stream>>>(
      dbc_w, dbcw_bf, (size_t)(R_ + 2 * N_) * E_);
  k_cvt_dtw<<<eb((size_t)E_ * 64), 256, 0, stream>>>(dt_w, dtw_bf);
  k_cvt<<<eb((size_t)H_ * E_), 256, 0, stream>>>(out_w, outw_bf, (size_t)H_ * E_);
  k_zero_guard<<<eb((size_t)B_ * 4 * E_), 256, 0, stream>>>(xyp_bf);

  // 3) in-projection: xy = x1 @ in_w^T + in_b (padded bf16) ; z = silu(xy)
  {
    const int tiles = (int)(M / 16) * ((E_ / 16) / 2);
    k_wmma_gemm<2, 0><<<(tiles + 7) / 8, 256, 0, stream>>>(
        x1_bf, H_, inw_bf, H_, (int)M, E_, H_, in_b, z, xyp_bf, nullptr);
  }

  // 4) per-direction: conv -> dbc -> delta -> scan
  for (int dir = 0; dir < 2; ++dir) {
    const unsigned short* cwT = dir ? bcwT : fcwT;
    const float* cb = dir ? bwd_cb : fwd_cb;
    const int tilesE2 = (int)(M / 16) * ((E_ / 16) / 2);
    k_wmma_conv<<<(tilesE2 + 7) / 8, 256, 0, stream>>>(xyp_bf, cwT, cb, dir, u_bf,
                                                       u_f);
    const int tilesDbc = (int)(M / 16) * ((R_ + 2 * N_) / 16);
    k_wmma_gemm<1, 2><<<(tilesDbc + 7) / 8, 256, 0, stream>>>(
        u_bf, E_, dbcw_bf, E_, (int)M, R_ + 2 * N_, E_, nullptr, dbc, nullptr,
        nullptr);
    k_cvt_dbc48<<<eb(M * 64), 256, 0, stream>>>(dbc, dbc48_bf);
    k_wmma_gemm<2, 3><<<(tilesE2 + 7) / 8, 256, 0, stream>>>(
        dbc48_bf, 64, dtw_bf, 64, (int)M, E_, 64, dt_b, delta, nullptr, nullptr);
    k_ssm<<<(B_ * E_ + 255) / 256, 256, 0, stream>>>(delta, dbc, u_f, A_log, Dp,
                                                     ysum, dir);
  }

  // 5) combine + out-projection with residual
  k_combine<<<eb(M * E_), 256, 0, stream>>>(ysum, z, s_bf, M * E_);
  {
    const int tiles = (int)(M / 16) * ((H_ / 16) / 2);
    k_wmma_gemm<2, 4><<<(tiles + 7) / 8, 256, 0, stream>>>(
        s_bf, E_, outw_bf, E_, (int)M, H_, E_, out_b, out, nullptr, x);
  }
}
